// SpatialMultiHeadAttention_82970178224629
// MI455X (gfx1250) — matrix operand order repaired and checked by reading, not compiled, not run on hardware
//
#include <hip/hip_runtime.h>
#include <hip/hip_bf16.h>

typedef __attribute__((ext_vector_type(16))) _Float16 v16h;
typedef __attribute__((ext_vector_type(8)))  _Float16 v8h;
typedef __attribute__((ext_vector_type(8)))  float    v8f;

#define B_TOK   16
#define N_TOK   1024
#define DIM_C   768
#define N_HEAD  12
#define HEAD_D  64
#define M_ROWS  (B_TOK * N_TOK)                                // 16384
#define QKV_ELEMS ((size_t)B_TOK * N_HEAD * N_TOK * HEAD_D)    // 12,582,912 per tensor

__device__ __forceinline__ v8f zero8() {
    v8f z;
#pragma unroll
    for (int i = 0; i < 8; ++i) z[i] = 0.0f;
    return z;
}

// Assemble a 16-half fragment from two contiguous 8-half chunks (contiguous k = 16*kh + s order, same as the B loads).
__device__ __forceinline__ v16h load_frag2(const _Float16* p0, const _Float16* p1) {
    v8h lo = *(const v8h*)p0;
    v8h hi = *(const v8h*)p1;
    v16h r;
#pragma unroll
    for (int i = 0; i < 8; ++i) { r[i] = lo[i]; r[i + 8] = hi[i]; }
    return r;
}

__device__ __forceinline__ v8f wmma_f16(v16h a, v16h b, v8f c) {
    return __builtin_amdgcn_wmma_f32_16x16x32_f16(false, a, false, b, (short)0, c,
                                                  false, false);
}

// ---------------------------------------------------------------------------
// Tiled WMMA GEMM:  C[M, Ncols] = A[M, K] * W[Ncols, K]^T + bias
//   EPI == 0 : f32 row-major store
//   EPI == 1 : f16 scatter into q/k/v head layout [3][B,H,N,64]
// Block: 256 threads (8 waves, 4x2 wave grid); block tile 128x128;
// wave tile 32x64. Double-buffered LDS, one barrier per 32-deep K-chunk.
// Staging converts f32->f16 in registers and stores b128 to LDS.
// ---------------------------------------------------------------------------
template <int EPI, typename AT>
__global__ __launch_bounds__(256) void gemm_wmma(const AT* __restrict__ A,
                                                 const float* __restrict__ W,
                                                 const float* __restrict__ bias,
                                                 void* __restrict__ dst,
                                                 int M, int Ncols, int K) {
    __shared__ __align__(64) _Float16 sA[2][128 * 32];
    __shared__ __align__(64) _Float16 sB[2][128 * 32];

    const int tid  = threadIdx.x;
    const int lane = tid & 31;
    const int wave = tid >> 5;
    const int waveM = wave & 3;
    const int waveN = wave >> 2;
    const int rowBase = blockIdx.x * 128;
    const int colBase = blockIdx.y * 128;

    const int mrow = lane & 15;
    const int kh   = lane >> 4;

    const int stR = tid >> 1;           // 0..127
    const int stC = (tid & 1) * 16;     // 0 or 16

    v8f acc[2][4];
#pragma unroll
    for (int mi = 0; mi < 2; ++mi)
#pragma unroll
        for (int ni = 0; ni < 4; ++ni) acc[mi][ni] = zero8();

    const int nChunks = K / 32;

    // ---- prologue: stage chunk 0 into buffer 0 ----
    {
        const AT* srcA = A + (size_t)(rowBase + stR) * K + stC;
        const float* srcB = W + (size_t)(colBase + stR) * K + stC;
        v8h a0, a1, b0, b1;
#pragma unroll
        for (int i = 0; i < 8; ++i) {
            a0[i] = (_Float16)(float)srcA[i];
            a1[i] = (_Float16)(float)srcA[8 + i];
            b0[i] = (_Float16)srcB[i];
            b1[i] = (_Float16)srcB[8 + i];
        }
        *(v8h*)&sA[0][stR * 32 + stC]     = a0;
        *(v8h*)&sA[0][stR * 32 + stC + 8] = a1;
        *(v8h*)&sB[0][stR * 32 + stC]     = b0;
        *(v8h*)&sB[0][stR * 32 + stC + 8] = b1;
    }

    for (int kt = 0; kt < nChunks; ++kt) {
        const int cur = kt & 1;
        __syncthreads();   // buffer `cur` staged; buffer `nxt` free to write

        if (kt + 1 < nChunks) {
            const int nxt = cur ^ 1;
            const int k0 = (kt + 1) * 32;
            const AT* srcA = A + (size_t)(rowBase + stR) * K + k0 + stC;
            const float* srcB = W + (size_t)(colBase + stR) * K + k0 + stC;
            v8h a0, a1, b0, b1;
#pragma unroll
            for (int i = 0; i < 8; ++i) {
                a0[i] = (_Float16)(float)srcA[i];
                a1[i] = (_Float16)(float)srcA[8 + i];
                b0[i] = (_Float16)srcB[i];
                b1[i] = (_Float16)srcB[8 + i];
            }
            *(v8h*)&sA[nxt][stR * 32 + stC]     = a0;
            *(v8h*)&sA[nxt][stR * 32 + stC + 8] = a1;
            *(v8h*)&sB[nxt][stR * 32 + stC]     = b0;
            *(v8h*)&sB[nxt][stR * 32 + stC + 8] = b1;
            if (kt + 2 < nChunks) {
                __builtin_prefetch(srcA + 32, 0, 1);
                __builtin_prefetch(srcB + 32, 0, 1);
            }
        }

        // ---- compute from buffer `cur` ----
        v16h afrag[2];
#pragma unroll
        for (int mi = 0; mi < 2; ++mi) {
            const _Float16* ap = &sA[cur][(waveM * 32 + mi * 16 + mrow) * 32];
            afrag[mi] = load_frag2(ap + 16 * kh, ap + 16 * kh + 8);
        }
#pragma unroll
        for (int ni = 0; ni < 4; ++ni) {
            const v16h bfrag =
                *(const v16h*)&sB[cur][(waveN * 64 + ni * 16 + mrow) * 32 + 16 * kh];
#pragma unroll
            for (int mi = 0; mi < 2; ++mi)
                acc[mi][ni] = wmma_f16(afrag[mi], bfrag, acc[mi][ni]);
        }
    }

    // ---- epilogue (C layout) ----
    const int rowHi = kh * 8;
#pragma unroll
    for (int mi = 0; mi < 2; ++mi) {
#pragma unroll
        for (int ni = 0; ni < 4; ++ni) {
            const int col = colBase + waveN * 64 + ni * 16 + mrow;
            const float bv = bias[col];
#pragma unroll
            for (int r = 0; r < 8; ++r) {
                const int row = rowBase + waveM * 32 + mi * 16 + rowHi + r;
                const float val = acc[mi][ni][r] + bv;
                if (EPI == 0) {
                    ((float*)dst)[(size_t)row * Ncols + col] = val;
                } else {
                    const int which = col / DIM_C;
                    const int rem   = col % DIM_C;
                    const int h     = rem / HEAD_D;
                    const int d     = rem % HEAD_D;
                    const int b     = row >> 10;
                    const int n     = row & 1023;
                    _Float16* q = (_Float16*)dst + (size_t)which * QKV_ELEMS +
                                  (((size_t)(b * N_HEAD + h)) * N_TOK + n) * HEAD_D + d;
                    *q = (_Float16)val;
                }
            }
        }
    }
}

// ---------------------------------------------------------------------------
// 2D RoPE applied in-place on a [B,H,N,64] f16 tensor, with an output scale
// (used to pre-fold the attention 1/sqrt(d) into q).
// ---------------------------------------------------------------------------
__global__ __launch_bounds__(256) void rope_kernel(_Float16* __restrict__ t,
                                                   float outScale) {
    const int idx = blockIdx.x * blockDim.x + threadIdx.x;
    const int j   = idx & 31;
    const int n   = (idx >> 5) & 1023;
    const int bh  = idx >> 15;
    if (bh >= B_TOK * N_HEAD) return;

    const int y = n >> 5, x = n & 31;
    const int fi = (j < 16) ? j : (j - 16);
    const float invf = __powf(10000.0f, -(float)fi / 16.0f);
    const float ang  = ((j < 16) ? (float)y : (float)x) * invf;
    const float c = __cosf(ang) * outScale, s = __sinf(ang) * outScale;
    const int d0 = (j < 16) ? (2 * j) : (32 + 2 * (j - 16));

    _Float16* p = t + (((size_t)bh << 10) + n) * HEAD_D + d0;
    const float t1 = (float)p[0], t2 = (float)p[1];
    p[0] = (_Float16)(t1 * c - t2 * s);
    p[1] = (_Float16)(t1 * s + t2 * c);
}

// ---------------------------------------------------------------------------
// Flash attention: block = one (b,h) x 128 query rows; wave = 16-row Q tile.
// 64-key blocks, double-buffered K/V staging (K row-major b128, V transposed
// with paired-key b32 packed stores). Row sums via ones-column WMMA; row max
// reduced with round-batched shuffles (8 bpermutes in flight per round).
// ---------------------------------------------------------------------------
__global__ __launch_bounds__(256) void attn_kernel(const _Float16* __restrict__ Q,
                                                   const _Float16* __restrict__ Kk,
                                                   const _Float16* __restrict__ V,
                                                   _Float16* __restrict__ O) {
    __shared__ __align__(64) _Float16 sK[2][64 * 64];     // [key][d]
    __shared__ __align__(64) _Float16 sVt[2][64 * 64];    // [d][key]
    __shared__ __align__(64) _Float16 sP[8][16 * 64];     // per-wave P tile

    const int tid  = threadIdx.x;
    const int lane = tid & 31;
    const int wave = tid >> 5;
    const int bh   = blockIdx.x >> 3;
    const int qblk = blockIdx.x & 7;
    const size_t base = (size_t)bh * N_TOK * HEAD_D;
    const int qrow0 = qblk * 128 + wave * 16;

    const int mrow = lane & 15;
    const int kh   = lane >> 4;

    // staging: thread owns a pair of keys x 8 d-values
    const int kp = (tid >> 3) * 2;      // keys kp, kp+1 (0..62)
    const int dc = (tid & 7) * 8;       // d 0..56

    // Preload Q fragments (two K=32 chunks covering d = 0..63)
    v16h qf[2];
    {
        const _Float16* qp = Q + base + (size_t)(qrow0 + mrow) * HEAD_D;
#pragma unroll
        for (int c = 0; c < 2; ++c)
            qf[c] = load_frag2(qp + 32 * c + 16 * kh, qp + 32 * c + 16 * kh + 8);
    }

    // Constant B-fragment of a "ones-column" tile: column n==0 is all ones.
    v16h onesfrag;
    {
        const _Float16 ov = (_Float16)((mrow == 0) ? 1.0f : 0.0f);
#pragma unroll
        for (int i = 0; i < 16; ++i) onesfrag[i] = ov;
    }

    v8f acc[4];
#pragma unroll
    for (int i = 0; i < 4; ++i) acc[i] = zero8();
    v8f accS = zero8();
    float rmax[8];
#pragma unroll
    for (int r = 0; r < 8; ++r) rmax[r] = -1e30f;

    const int NB = N_TOK / 64;          // 16 key blocks

    // ---- prologue: stage key block 0 into buffer 0 ----
    {
        const _Float16* srcK = Kk + base + (size_t)kp * HEAD_D + dc;
        const _Float16* srcV = V  + base + (size_t)kp * HEAD_D + dc;
        *(v8h*)&sK[0][kp * 64 + dc]       = *(const v8h*)srcK;
        *(v8h*)&sK[0][(kp + 1) * 64 + dc] = *(const v8h*)(srcK + HEAD_D);
        const v8h v0 = *(const v8h*)srcV;
        const v8h v1 = *(const v8h*)(srcV + HEAD_D);
#pragma unroll
        for (int i = 0; i < 8; ++i) {
            union { _Float16 h[2]; unsigned u; } pk;
            pk.h[0] = v0[i]; pk.h[1] = v1[i];
            *(unsigned*)&sVt[0][(dc + i) * 64 + kp] = pk.u;
        }
    }

    for (int kb = 0; kb < NB; ++kb) {
        const int cur = kb & 1;
        __syncthreads();

        if (kb + 1 < NB) {
            const int nxt = cur ^ 1;
            const _Float16* srcK =
                Kk + base + (size_t)((kb + 1) * 64 + kp) * HEAD_D + dc;
            const _Float16* srcV =
                V + base + (size_t)((kb + 1) * 64 + kp) * HEAD_D + dc;
            *(v8h*)&sK[nxt][kp * 64 + dc]       = *(const v8h*)srcK;
            *(v8h*)&sK[nxt][(kp + 1) * 64 + dc] = *(const v8h*)(srcK + HEAD_D);
            const v8h v0 = *(const v8h*)srcV;
            const v8h v1 = *(const v8h*)(srcV + HEAD_D);
#pragma unroll
            for (int i = 0; i < 8; ++i) {
                union { _Float16 h[2]; unsigned u; } pk;
                pk.h[0] = v0[i]; pk.h[1] = v1[i];
                *(unsigned*)&sVt[nxt][(dc + i) * 64 + kp] = pk.u;
            }
            if (kb + 2 < NB) __builtin_prefetch(srcK + 64 * HEAD_D, 0, 1);
        }

        // ---- S = Q K^T (pre-scaled) for four 16-key tiles ----
        v8f s[4];
#pragma unroll
        for (int t = 0; t < 4; ++t) s[t] = zero8();
#pragma unroll
        for (int c = 0; c < 2; ++c) {
#pragma unroll
            for (int t = 0; t < 4; ++t) {
                const v16h bf =
                    *(const v16h*)&sK[cur][(t * 16 + mrow) * 64 + 32 * c + 16 * kh];
                s[t] = wmma_f16(qf[c], bf, s[t]);
            }
        }

        // ---- row max: round-batched butterfly (8 permutes in flight) ----
        float vmax[8];
#pragma unroll
        for (int r = 0; r < 8; ++r)
            vmax[r] = fmaxf(fmaxf(s[0][r], s[1][r]), fmaxf(s[2][r], s[3][r]));
#pragma unroll
        for (int off = 8; off >= 1; off >>= 1) {
            float t[8];
#pragma unroll
            for (int r = 0; r < 8; ++r) t[r] = __shfl_xor(vmax[r], off, 32);
#pragma unroll
            for (int r = 0; r < 8; ++r) vmax[r] = fmaxf(vmax[r], t[r]);
        }

        // ---- online softmax update ----
        float alpha[8];
#pragma unroll
        for (int r = 0; r < 8; ++r) {
            const float nm = fmaxf(rmax[r], vmax[r]);
            alpha[r] = __expf(rmax[r] - nm);
            rmax[r] = nm;
#pragma unroll
            for (int t = 0; t < 4; ++t) s[t][r] = __expf(s[t][r] - nm);
        }
#pragma unroll
        for (int nt = 0; nt < 4; ++nt)
#pragma unroll
            for (int r = 0; r < 8; ++r) acc[nt][r] *= alpha[r];
#pragma unroll
        for (int r = 0; r < 8; ++r) accS[r] *= alpha[r];

        // ---- re-layout P (C layout -> A layout) via per-wave LDS tile ----
        _Float16* pw = &sP[wave][0];
#pragma unroll
        for (int r = 0; r < 8; ++r) {
            const int row = r + 8 * kh;
#pragma unroll
            for (int t = 0; t < 4; ++t)
                pw[row * 64 + t * 16 + mrow] = (_Float16)s[t][r];
        }
        const _Float16* pr = &pw[mrow * 64];
        v16h pf[2];
#pragma unroll
        for (int c = 0; c < 2; ++c)
            pf[c] = load_frag2(pr + 32 * c + 16 * kh, pr + 32 * c + 16 * kh + 8);

        // ---- [O | l] += P [V | 1] ----
#pragma unroll
        for (int c = 0; c < 2; ++c) {
#pragma unroll
            for (int nt = 0; nt < 4; ++nt) {
                const v16h vf =
                    *(const v16h*)&sVt[cur][(nt * 16 + mrow) * 64 + 32 * c + 16 * kh];
                acc[nt] = wmma_f16(pf[c], vf, acc[nt]);
            }
            accS = wmma_f16(pf[c], onesfrag, accS);
        }
    }

    // ---- normalize + store into [B, N, 768] with column = h*64 + d ----
    const int hh = bh % N_HEAD, bb = bh / N_HEAD;
    const int bcast = kh * 16;          // lane holding column 0 of this half
#pragma unroll
    for (int r = 0; r < 8; ++r) {
        const int row = qrow0 + 8 * kh + r;
        const float l = __shfl(accS[r], bcast, 32);
        const float inv = 1.0f / l;
#pragma unroll
        for (int nt = 0; nt < 4; ++nt) {
            const int d = nt * 16 + mrow;
            O[((size_t)bb * N_TOK + row) * DIM_C + hh * HEAD_D + d] =
                (_Float16)(acc[nt][r] * inv);
        }
    }
}

extern "C" void kernel_launch(void* const* d_in, const int* in_sizes, int n_in,
                              void* d_out, int out_size, void* d_ws, size_t ws_size,
                              hipStream_t stream) {
    const float* x      = (const float*)d_in[0];
    const float* qkv_w  = (const float*)d_in[1];
    const float* qkv_b  = (const float*)d_in[2];
    const float* proj_w = (const float*)d_in[3];
    const float* proj_b = (const float*)d_in[4];
    (void)in_sizes; (void)n_in; (void)out_size; (void)ws_size;

    _Float16* qkvh = (_Float16*)d_ws;                         // [3][B,H,N,64] f16
    _Float16* attn = (_Float16*)((char*)d_ws + 3 * QKV_ELEMS * sizeof(_Float16));
    float* out = (float*)d_out;

    // 1) QKV GEMM (+bias), scatter into head layout as f16
    dim3 g1(M_ROWS / 128, (3 * DIM_C) / 128);   // 128 x 18
    gemm_wmma<1, float><<<g1, 256, 0, stream>>>(x, qkv_w, qkv_b, qkvh,
                                                M_ROWS, 3 * DIM_C, DIM_C);
    // 2) 2D RoPE on q (folding in 1/sqrt(d)) and k (in place)
    const int ropeThreads = B_TOK * N_HEAD * N_TOK * 32;
    rope_kernel<<<ropeThreads / 256, 256, 0, stream>>>(qkvh, 0.125f);
    rope_kernel<<<ropeThreads / 256, 256, 0, stream>>>(qkvh + QKV_ELEMS, 1.0f);
    // 3) Flash attention -> [B, N, 768] f16
    attn_kernel<<<B_TOK * N_HEAD * 8, 256, 0, stream>>>(
        qkvh, qkvh + QKV_ELEMS, qkvh + 2 * QKV_ELEMS, attn);
    // 4) Output projection (+bias) -> f32 d_out
    dim3 g2(M_ROWS / 128, DIM_C / 128);         // 128 x 6
    gemm_wmma<0, _Float16><<<g2, 256, 0, stream>>>(attn, proj_w, proj_b, out,
                                                   M_ROWS, DIM_C, DIM_C);
}
